// MultiHeadAttentionWithRoPE_11836929868207
// MI455X (gfx1250) — matrix operand order repaired and checked by reading, not compiled, not run on hardware
//
#include <hip/hip_runtime.h>
#include <hip/hip_bf16.h>

// MultiHeadAttentionWithRoPE for MI455X (gfx1250), bf16 WMMA pipeline.
// B=2, S=2048, D=1024, H=16, dk=64. fp32 in/out, bf16 matrix operands,
// f32 accumulation via V_WMMA_F32_16X16X32_BF16.
// Flash kernel stages K tiles into LDS with GLOBAL_LOAD_ASYNC_TO_LDS_B128
// (ASYNCcnt path) and relays P through LDS for the PV WMMA.

typedef __attribute__((ext_vector_type(16))) __bf16 v16bf;
typedef __attribute__((ext_vector_type(8)))  __bf16 v8bf;
typedef __attribute__((ext_vector_type(8)))  float  v8f;

#define BATCH 2
#define SEQ   2048
#define DMODEL 1024
#define NHEAD 16
#define DK    64
#define MROWS (BATCH * SEQ)         // 4096
#define LN10000_OVER_32 0.2878231366242536f

// ---------------------------------------------------------------- helpers

__device__ __forceinline__ v8f vzero8() {
  v8f z;
#pragma unroll
  for (int i = 0; i < 8; ++i) z[i] = 0.0f;
  return z;
}

__device__ __forceinline__ v8f wmma_bf16(v16bf a, v16bf b, v8f c) {
  // D = A(16x32) * B(32x16) + C, f32 accumulate
  return __builtin_amdgcn_wmma_f32_16x16x32_bf16(
      /*neg_a=*/false, a, /*neg_b=*/false, b,
      /*c_mod=*/(short)0, c, /*reuse_a=*/false, /*reuse_b=*/false);
}

// A fragment 16x32 bf16 from row-major [16 x ld] at (row 0..15, k0..k0+31).
// Layout (ISA 7.12.2): lanes 0-15 hold M=lane, K = {0..7, 16..23};
//                      lanes 16-31 hold M=lane-16, K = {8..15, 24..31}.
__device__ __forceinline__ v16bf load_a_frag(const __bf16* rowbase, int ld,
                                             int k0, int lane) {
  const int m = lane & 15, hi = lane >> 4;
  const __bf16* p = rowbase + (size_t)m * ld + k0 + hi * 8;
  v8bf lo = *(const v8bf*)(p);
  v8bf up = *(const v8bf*)(p + 16);
  v16bf r;
#pragma unroll
  for (int i = 0; i < 8; ++i) { r[i] = lo[i]; r[i + 8] = up[i]; }
  return r;
}

// B fragment 32x16 bf16 where B^T is row-major [N x ld] (K contiguous).
// Layout: lanes 0-15 hold N=lane, K={0..7,16..23}; lanes 16-31 hold N=lane-16, K={8..15,24..31}.
__device__ __forceinline__ v16bf load_b_frag(const __bf16* bt, int ld, int n0,
                                             int k0, int lane) {
  const int n = lane & 15, hi = lane >> 4;
  const __bf16* p = bt + (size_t)(n0 + n) * ld + k0 + hi * 8;
  v8bf lo = *(const v8bf*)(p);
  v8bf up = *(const v8bf*)(p + 16);
  v16bf r;
#pragma unroll
  for (int i = 0; i < 8; ++i) { r[i] = lo[i]; r[i + 8] = up[i]; }
  return r;
}

// A fragment 16x32 from an LDS tile stored row-major 16x32 bf16.
__device__ __forceinline__ v16bf load_p_frag(const __bf16* lp, int lane) {
  const int m = lane & 15, hi = lane >> 4;
  v8bf lo = *(const v8bf*)(lp + m * 32 + hi * 8);
  v8bf up = *(const v8bf*)(lp + m * 32 + 16 + hi * 8);
  v16bf r;
#pragma unroll
  for (int i = 0; i < 8; ++i) { r[i] = lo[i]; r[i + 8] = up[i]; }
  return r;
}

// Generic LDS pointer -> LDS byte offset. Per ISA 10.2 the LDS aperture maps
// addr[31:0] directly to the LDS offset, so the low 32 bits are the offset.
__device__ __forceinline__ unsigned lds_off_of(const void* p) {
  return (unsigned)(unsigned long long)p;
}

// Async copy 16 bytes/lane from global to LDS (GLOBAL_LOAD_ASYNC_TO_LDS_B128,
// GV mode, tracked by ASYNCcnt).
__device__ __forceinline__ void async_b128_to_lds(unsigned lds_byte_off,
                                                  const void* gptr) {
  unsigned long long ga = (unsigned long long)gptr;
  asm volatile("global_load_async_to_lds_b128 %0, %1, off"
               :: "v"(lds_byte_off), "v"(ga) : "memory");
}

__device__ __forceinline__ void wait_asynccnt0() {
  asm volatile("s_wait_asynccnt 0x0" ::: "memory");
}

// ---------------------------------------------------------------- kernels

__global__ __launch_bounds__(256) void cvt_f32_bf16_kernel(
    const float* __restrict__ in, __bf16* __restrict__ out, int n) {
  int i = blockIdx.x * blockDim.x + threadIdx.x;
  if (i < n) out[i] = (__bf16)in[i];
}

// Epilogue for one 16x64 row-group of the projection GEMM.
__device__ __forceinline__ void proj_epilogue(
    int mode, int m0, int n0, int hd, int nlo, int hi,
    const v8f acc[4], const float* __restrict__ bias,
    __bf16* __restrict__ Qr, __bf16* __restrict__ Kr,
    __bf16* __restrict__ Vt, float* __restrict__ dout) {
  v8f vals[4];
#pragma unroll
  for (int f = 0; f < 4; ++f) {
    float bb = bias[n0 + f * 16 + nlo];
#pragma unroll
    for (int r = 0; r < 8; ++r) vals[f][r] = acc[f][r] + bb;
  }
  if (mode == 0) {
#pragma unroll
    for (int f = 0; f < 4; ++f)
#pragma unroll
      for (int r = 0; r < 8; ++r) vals[f][r] *= 0.125f;  // 1/sqrt(64)
  }

  if (mode <= 1) {
    // RoPE: out[i] = v[i]*cos(s*invf[i>>1]) + rot(v)[i]*sin(...),
    // rot(v)[i] = (i<32) ? -v[i+32] : v[i-32]  (pairing across frag f <-> f^2)
    __bf16* dst = (mode == 0) ? Qr : Kr;
    v8f outv[4];
#pragma unroll
    for (int f = 0; f < 4; ++f) {
      const int i = f * 16 + nlo;  // 0..63 within head
      const float invf = __expf(-LN10000_OVER_32 * (float)(i >> 1));
#pragma unroll
      for (int r = 0; r < 8; ++r) {
        const int m = m0 + r + hi * 8;
        const float ang = (float)(m & (SEQ - 1)) * invf;
        float sv, cv;
        __sincosf(ang, &sv, &cv);
        const float partner = (f < 2) ? -vals[f + 2][r] : vals[f - 2][r];
        outv[f][r] = vals[f][r] * cv + partner * sv;
      }
    }
#pragma unroll
    for (int f = 0; f < 4; ++f) {
      const int i = f * 16 + nlo;
#pragma unroll
      for (int r = 0; r < 8; ++r) {
        const int m = m0 + r + hi * 8;
        const int b = m >> 11;           // token row -> batch
        const int s = m & (SEQ - 1);
        dst[((size_t)(b * NHEAD + hd) * SEQ + s) * DK + i] = (__bf16)outv[f][r];
      }
    }
  } else if (mode == 2) {
#pragma unroll
    for (int f = 0; f < 4; ++f) {
      const int i = f * 16 + nlo;
#pragma unroll
      for (int r = 0; r < 8; ++r) {
        const int m = m0 + r + hi * 8;
        const int b = m >> 11;
        const int s = m & (SEQ - 1);
        Vt[((size_t)(b * NHEAD + hd) * DK + i) * SEQ + s] = (__bf16)vals[f][r];
      }
    }
  } else {
#pragma unroll
    for (int f = 0; f < 4; ++f)
#pragma unroll
      for (int r = 0; r < 8; ++r) {
        const int m = m0 + r + hi * 8;
        dout[(size_t)m * DMODEL + n0 + f * 16 + nlo] = vals[f][r];
      }
  }
}

// Fused projection GEMM. mode = mode_base + blockIdx.z:
//   0: Q = x@wq^T + bq, scale 1/sqrt(dk), RoPE, -> Qr [B,H,S,DK] bf16
//   1: K = x@wk^T + bk, RoPE,                   -> Kr [B,H,S,DK] bf16
//   2: V = x@wv^T + bv,                         -> Vt [B,H,DK,S] bf16 (transposed)
//   3: out = attn@wo^T + bo                     -> dout [B,S,D] f32
// Each wave computes a 32x64 tile (two 16-row groups share every B fragment,
// full head width so RoPE's i<->i+32 partner sits two fragments over).
__global__ __launch_bounds__(256) void proj_gemm_kernel(
    const __bf16* __restrict__ xb, const __bf16* __restrict__ attnb_in,
    const __bf16* __restrict__ wqb, const __bf16* __restrict__ wkb,
    const __bf16* __restrict__ wvb, const __bf16* __restrict__ wob,
    const float* __restrict__ bq, const float* __restrict__ bk,
    const float* __restrict__ bv, const float* __restrict__ bo,
    __bf16* __restrict__ Qr, __bf16* __restrict__ Kr,
    __bf16* __restrict__ Vt, float* __restrict__ dout, int mode_base) {
  const int mode = mode_base + blockIdx.z;
  const int lane = threadIdx.x & 31;
  const int w    = threadIdx.x >> 5;
  const int nlo  = lane & 15;
  const int hi   = lane >> 4;
  const int m0   = blockIdx.x * 256 + w * 32;   // output row tile (token rows)
  const int n0   = blockIdx.y * 64;             // output col tile (head-aligned)
  const int hd   = blockIdx.y;

  const __bf16* A    = (mode < 3) ? xb : attnb_in;
  const __bf16* Bt   = (mode == 0) ? wqb : (mode == 1) ? wkb
                     : (mode == 2) ? wvb : wob;
  const float*  bias = (mode == 0) ? bq : (mode == 1) ? bk
                     : (mode == 2) ? bv : bo;
  const __bf16* arow0 = A + (size_t)m0 * DMODEL;
  const __bf16* arow1 = arow0 + (size_t)16 * DMODEL;

  v8f acc[2][4];
#pragma unroll
  for (int g = 0; g < 2; ++g)
#pragma unroll
    for (int f = 0; f < 4; ++f) acc[g][f] = vzero8();

  for (int k0 = 0; k0 < DMODEL; k0 += 32) {
    v16bf a0 = load_a_frag(arow0, DMODEL, k0, lane);
    v16bf a1 = load_a_frag(arow1, DMODEL, k0, lane);
#pragma unroll
    for (int f = 0; f < 4; ++f) {
      v16bf b = load_b_frag(Bt, DMODEL, n0 + f * 16, k0, lane);
      acc[0][f] = wmma_bf16(a0, b, acc[0][f]);
      acc[1][f] = wmma_bf16(a1, b, acc[1][f]);
    }
  }

#pragma unroll
  for (int g = 0; g < 2; ++g)
    proj_epilogue(mode, m0 + g * 16, n0, hd, nlo, hi, acc[g], bias,
                  Qr, Kr, Vt, dout);
}

// Flash-style causal attention. Grid: (S/128, B*H). 8 waves/block; each wave
// owns 16 query rows, streams KV in 32-wide tiles with online softmax.
// K tiles are staged into LDS with async global->LDS copies (ASYNCcnt);
// P (f32 C-frag) is relaid to a bf16 A-frag through a wave-private LDS tile.
__global__ __launch_bounds__(256) void flash_attn_kernel(
    const __bf16* __restrict__ Qr, const __bf16* __restrict__ Kr,
    const __bf16* __restrict__ Vt, __bf16* __restrict__ attnb) {
  __shared__ __align__(64) __bf16 ldsK[8][32 * 64];  // 32 KB, K tile per wave
  __shared__ __align__(64) __bf16 ldsP[8][16 * 32];  // 8 KB, P tile per wave

  const int lane = threadIdx.x & 31;
  const int w    = threadIdx.x >> 5;
  const int nlo  = lane & 15;
  const int hi   = lane >> 4;
  const int bh   = blockIdx.y;
  const int b    = bh >> 4;
  const int hd   = bh & 15;
  const int q0   = blockIdx.x * 128 + w * 16;

  const __bf16* Qb = Qr + (size_t)bh * SEQ * DK;  // [S, DK]
  const __bf16* Kb = Kr + (size_t)bh * SEQ * DK;  // [S, DK]
  const __bf16* Vb = Vt + (size_t)bh * DK * SEQ;  // [DK, S] (transposed)

  const __bf16* kt = &ldsK[w][0];
  const unsigned ktOff = lds_off_of(kt);

  // Q row-panel: 16 x 64 -> two A fragments (K = 0..31, 32..63)
  const v16bf aq0 = load_a_frag(Qb + (size_t)q0 * DK, DK, 0, lane);
  const v16bf aq1 = load_a_frag(Qb + (size_t)q0 * DK, DK, 32, lane);

  v8f acc[4];
#pragma unroll
  for (int f = 0; f < 4; ++f) acc[f] = vzero8();
  float m_i[8], l_i[8];
#pragma unroll
  for (int r = 0; r < 8; ++r) { m_i[r] = -3.0e38f; l_i[r] = 0.0f; }

  const int numT = (q0 + 15) / 32 + 1;  // causal: only tiles with kv <= q
  for (int t = 0; t < numT; ++t) {
    const int j2 = t * 32;

    // Stage K tile (32 rows x 64 bf16 = contiguous 4 KB) into LDS:
    // 8 async b128 issues x 32 lanes x 16 B.
    {
      const char* g = (const char*)(Kb + (size_t)j2 * DK);
#pragma unroll
      for (int i = 0; i < 8; ++i) {
        const int c = i * 32 + lane;               // 16-byte chunk index
        async_b128_to_lds(ktOff + (unsigned)(c * 16), g + (size_t)c * 16);
      }
      wait_asynccnt0();
    }

    // scores S = Q * K^T : two 16x16 f32 fragments (kv cols j2.., j2+16..),
    // B fragments read from the LDS-staged K tile.
    v8f s0 = vzero8(), s1 = vzero8();
    s0 = wmma_bf16(aq0, load_b_frag(kt, DK, 0, 0, lane), s0);
    s0 = wmma_bf16(aq1, load_b_frag(kt, DK, 0, 32, lane), s0);
    s1 = wmma_bf16(aq0, load_b_frag(kt, DK, 16, 0, lane), s1);
    s1 = wmma_bf16(aq1, load_b_frag(kt, DK, 16, 32, lane), s1);

    // causal mask + online softmax (rows live in vgpr index + lane half)
#pragma unroll
    for (int r = 0; r < 8; ++r) {
      const int qrow = q0 + r + hi * 8;
      const float v0 = (j2 + nlo <= qrow) ? s0[r] : -1.0e9f;
      const float v1 = (j2 + 16 + nlo <= qrow) ? s1[r] : -1.0e9f;
      float mx = fmaxf(v0, v1);
#pragma unroll
      for (int msk = 1; msk < 16; msk <<= 1) mx = fmaxf(mx, __shfl_xor(mx, msk, 32));
      const float mnew = fmaxf(m_i[r], mx);
      const float sc   = __expf(m_i[r] - mnew);
      const float p0   = __expf(v0 - mnew);
      const float p1   = __expf(v1 - mnew);
      float ps = p0 + p1;
#pragma unroll
      for (int msk = 1; msk < 16; msk <<= 1) ps += __shfl_xor(ps, msk, 32);
      l_i[r] = l_i[r] * sc + ps;
      m_i[r] = mnew;
#pragma unroll
      for (int f = 0; f < 4; ++f) acc[f][r] *= sc;
      const int row = r + hi * 8;
      ldsP[w][row * 32 + nlo]      = (__bf16)p0;
      ldsP[w][row * 32 + 16 + nlo] = (__bf16)p1;
    }

    // relay P through LDS into A-frag layout (same wave: DS ops are in-order)
    const v16bf pA = load_p_frag(&ldsP[w][0], lane);

    // acc += P(16x32) * V(32x64): 4 WMMAs over dk tiles, B from transposed V
#pragma unroll
    for (int f = 0; f < 4; ++f)
      acc[f] = wmma_bf16(pA, load_b_frag(Vb, SEQ, f * 16, j2, lane), acc[f]);
  }

  // normalize and scatter to attn buffer [B,S,D] bf16
#pragma unroll
  for (int r = 0; r < 8; ++r) {
    const float rl = 1.0f / l_i[r];
    const int s = q0 + r + hi * 8;
#pragma unroll
    for (int f = 0; f < 4; ++f) {
      const float v = acc[f][r] * rl;
      attnb[((size_t)(b * SEQ + s)) * DMODEL + hd * DK + f * 16 + nlo] = (__bf16)v;
    }
  }
}

// ---------------------------------------------------------------- launcher

extern "C" void kernel_launch(void* const* d_in, const int* in_sizes, int n_in,
                              void* d_out, int out_size, void* d_ws, size_t ws_size,
                              hipStream_t stream) {
  (void)in_sizes; (void)n_in; (void)out_size; (void)ws_size;
  const float* x  = (const float*)d_in[0];
  // d_in[1] = mask (causal tril) — implied by kernel, not read
  const float* wq = (const float*)d_in[2];
  const float* bq = (const float*)d_in[3];
  const float* wk = (const float*)d_in[4];
  const float* bk = (const float*)d_in[5];
  const float* wv = (const float*)d_in[6];
  const float* bv = (const float*)d_in[7];
  const float* wo = (const float*)d_in[8];
  const float* bo = (const float*)d_in[9];
  float* dout = (float*)d_out;

  // workspace layout (48 MiB total)
  unsigned char* ws = (unsigned char*)d_ws;
  __bf16* xb    = (__bf16*)(ws);                           // 8 MiB
  __bf16* wqb   = (__bf16*)(ws + ( 8u << 20));             // 2 MiB
  __bf16* wkb   = (__bf16*)(ws + (10u << 20));             // 2 MiB
  __bf16* wvb   = (__bf16*)(ws + (12u << 20));             // 2 MiB
  __bf16* wob   = (__bf16*)(ws + (14u << 20));             // 2 MiB
  __bf16* Qr    = (__bf16*)(ws + (16u << 20));             // 8 MiB [B,H,S,DK]
  __bf16* Kr    = (__bf16*)(ws + (24u << 20));             // 8 MiB [B,H,S,DK]
  __bf16* Vt    = (__bf16*)(ws + (32u << 20));             // 8 MiB [B,H,DK,S]
  __bf16* attnb = (__bf16*)(ws + (40u << 20));             // 8 MiB [B,S,D]

  const int NX = MROWS * DMODEL;    // 4194304
  const int NW = DMODEL * DMODEL;   // 1048576
  cvt_f32_bf16_kernel<<<NX / 256, 256, 0, stream>>>(x,  xb,  NX);
  cvt_f32_bf16_kernel<<<NW / 256, 256, 0, stream>>>(wq, wqb, NW);
  cvt_f32_bf16_kernel<<<NW / 256, 256, 0, stream>>>(wk, wkb, NW);
  cvt_f32_bf16_kernel<<<NW / 256, 256, 0, stream>>>(wv, wvb, NW);
  cvt_f32_bf16_kernel<<<NW / 256, 256, 0, stream>>>(wo, wob, NW);

  // fused QKV projection + bias + scale + RoPE (+ V transpose)
  proj_gemm_kernel<<<dim3(MROWS / 256, DMODEL / 64, 3), 256, 0, stream>>>(
      xb, attnb, wqb, wkb, wvb, wob, bq, bk, bv, bo, Qr, Kr, Vt, dout, 0);

  // causal flash attention (async K staging into LDS)
  flash_attn_kernel<<<dim3(SEQ / 128, BATCH * NHEAD), 256, 0, stream>>>(
      Qr, Kr, Vt, attnb);

  // output projection -> f32 d_out
  proj_gemm_kernel<<<dim3(MROWS / 256, DMODEL / 64, 1), 256, 0, stream>>>(
      xb, attnb, wqb, wkb, wvb, wob, bq, bk, bv, bo, Qr, Kr, Vt, dout, 3);
}
